// ResidualGCN_12893491822680
// MI455X (gfx1250) — matrix operand order repaired and checked by reading, not compiled, not run on hardware
//
#include <hip/hip_runtime.h>

#define DEV __device__ __forceinline__

typedef __attribute__((ext_vector_type(16))) __bf16          v16bf;
typedef __attribute__((ext_vector_type(16))) float           v16f;
typedef __attribute__((ext_vector_type(8)))  float           v8f;

static constexpr int Hdim = 128;
static constexpr float LN_EPS = 1e-5f;

DEV unsigned short f2bf(float f) {
  unsigned u = __builtin_bit_cast(unsigned, f);
  u += 0x7FFFu + ((u >> 16) & 1u);            // round-to-nearest-even
  return (unsigned short)(u >> 16);
}

// ---------------------------------------------------------------------------
// WMMA GEMM: C[M,NOUT] = act(A[M,K] @ W[K,NOUT] + bias)
// bf16 operands, fp32 accumulate. W staged once into LDS pre-swizzled into the
// CDNA5 B-fragment layout; A fragments register-resident across all of K.
//   KC = K/32 k-chunks, NT = NOUT/16 column tiles.
// Block = 256 threads (8 waves), each block covers 128 rows.
// ---------------------------------------------------------------------------
template <int KC, int NT, bool RELU, bool BIAS>
__global__ __launch_bounds__(256)
void gemm_bf16_wmma(const float* __restrict__ A, const float* __restrict__ W,
                    const float* __restrict__ bias, float* __restrict__ C, int M) {
  constexpr int K    = KC * 32;
  constexpr int NOUT = NT * 16;
  extern __shared__ unsigned short sW[];   // [NT][KC][32 lanes][16 bf16]

  // Stage W (row-major fp32 [K,NOUT]) into LDS in B-fragment order:
  // B 32x16 bf16 (ISA 7.12.2): lanes 0-15 hold K {0..7,16..23}, lanes 16-31 hold K {8..15,24..31},
  // slots 0-7 hold the low K run, slots 8-15 the run 16 higher (same as A).
  for (int idx = threadIdx.x; idx < K * NOUT; idx += 256) {
    int k = idx / NOUT, n = idx % NOUT;
    int kc   = k >> 5;
    int r    = k & 31;
    int lane = (((r >> 3) & 1) << 4) | (n & 15);
    int p    = (r & 7) | ((r >> 4) << 3);
    int nt   = n >> 4;
    sW[((((nt * KC + kc) << 5) + lane) << 4) + p] = f2bf(W[idx]);
  }
  __syncthreads();

  const int wave    = threadIdx.x >> 5;
  const int lane    = threadIdx.x & 31;
  const int rowBase = blockIdx.x * 128 + wave * 16;
  int rowA = rowBase + (lane & 15);
  if (rowA > M - 1) rowA = M - 1;          // clamp loads; stores guarded below

  // A fragments (16x32 bf16, ISA 7.12.2): lanes 0-15 carry K {0..7,16..23},
  // lanes 16-31 carry K {8..15,24..31}. Register-resident for the whole K loop.
  // fp32 -> bf16 via vector convert so the backend can use packed cvt ops.
  v16bf a[KC];
  const int kb = (lane >> 4) * 8;
#pragma unroll
  for (int kc = 0; kc < KC; ++kc) {
    const float* ap = A + (size_t)rowA * K + kc * 32 + kb;
    v8f lo = *(const v8f*)(ap);        // K +0..7   (32B aligned)
    v8f hi = *(const v8f*)(ap + 16);   // K +16..23 (32B aligned)
    v16f af = __builtin_shufflevector(lo, hi, 0, 1, 2, 3, 4, 5, 6, 7,
                                      8, 9, 10, 11, 12, 13, 14, 15);
    a[kc] = __builtin_convertvector(af, v16bf);
  }

#pragma unroll
  for (int nt = 0; nt < NT; ++nt) {
    v8f acc = {};
#pragma unroll
    for (int kc = 0; kc < KC; ++kc) {
      const v16bf b =
          *(const v16bf*)(&sW[((((nt * KC + kc) << 5) + lane) << 4)]);
      acc = __builtin_amdgcn_wmma_f32_16x16x32_bf16(
          /*neg_a=*/false, a[kc], /*neg_b=*/false, b,
          /*c_mod=*/(short)0, acc, /*reuse_a=*/false, /*reuse_b=*/false);
    }
    const int col = nt * 16 + (lane & 15);
    float bv = 0.f;
    if (BIAS) bv = bias[col];
    // C/D 16x16 f32 layout: lanes 0-15 -> rows v, lanes 16-31 -> rows v+8.
#pragma unroll
    for (int v = 0; v < 8; ++v) {
      int gr = rowBase + ((lane < 16) ? v : v + 8);
      if (gr < M) {
        float o = acc[v] + bv;
        if (RELU) o = fmaxf(o, 0.f);
        C[(size_t)gr * NOUT + col] = o;
      }
    }
  }
}

// ---------------------------------------------------------------------------
// Graph plumbing kernels
// ---------------------------------------------------------------------------
__global__ void k_fill0(float* __restrict__ p, int n) {
  int i = blockIdx.x * 256 + threadIdx.x;
  if (i < n) p[i] = 0.f;
}

__global__ void k_degree(const long long* __restrict__ ei, float* __restrict__ deg,
                         int E_) {
  int e = blockIdx.x * 256 + threadIdx.x;
  if (e < E_) atomicAdd(&deg[(int)ei[E_ + e]], 1.0f);
}

__global__ void k_rsqrt(float* __restrict__ p, int n) {
  int i = blockIdx.x * 256 + threadIdx.x;
  if (i < n) p[i] = rsqrtf(p[i] + 1.0f);   // dis = rsqrt(deg + 1)
}

// agg = m * (dis*dis) + conv_b   (self-loop + bias; also initializes agg)
__global__ void k_selfbias(const float* __restrict__ m, const float* __restrict__ dis,
                           const float* __restrict__ cb, float* __restrict__ agg,
                           int total) {
  int i = blockIdx.x * 256 + threadIdx.x;
  if (i < total) {
    int node = i >> 7, f = i & 127;
    float d = dis[node];
    agg[i] = m[i] * d * d + cb[f];
  }
}

// one wave32 per edge: agg[dst] += m[src] * (dis[src]*dis[dst])
__global__ __launch_bounds__(256)
void k_scatter(const long long* __restrict__ ei, const float* __restrict__ dis,
               const float* __restrict__ m, float* __restrict__ agg, int E_) {
  int e = blockIdx.x * 8 + (threadIdx.x >> 5);
  if (e >= E_) return;
  int lane = threadIdx.x & 31;
  int s = (int)ei[e];
  int d = (int)ei[E_ + e];
  float nrm = dis[s] * dis[d];
  const float4 v = *(const float4*)(m + (size_t)s * Hdim + lane * 4);
  float* ap = agg + (size_t)d * Hdim + lane * 4;
  atomicAdd(ap + 0, v.x * nrm);
  atomicAdd(ap + 1, v.y * nrm);
  atomicAdd(ap + 2, v.z * nrm);
  atomicAdd(ap + 3, v.w * nrm);
}

// one wave32 per node: hnew = relu(layernorm(agg)*g + b) + hold
__global__ __launch_bounds__(256)
void k_ln_relu_res(const float* __restrict__ agg, const float* __restrict__ g,
                   const float* __restrict__ bb, const float* __restrict__ hold,
                   float* __restrict__ hnew, int n) {
  int node = blockIdx.x * 8 + (threadIdx.x >> 5);
  if (node >= n) return;
  int lane = threadIdx.x & 31;
  size_t base = (size_t)node * Hdim + lane * 4;
  float4 a = *(const float4*)(agg + base);
  float s = a.x + a.y + a.z + a.w;
#pragma unroll
  for (int o = 16; o; o >>= 1) s += __shfl_xor(s, o, 32);
  float mu = s * (1.0f / 128.0f);
  float dx = a.x - mu, dy = a.y - mu, dz = a.z - mu, dw = a.w - mu;
  float vs = dx * dx + dy * dy + dz * dz + dw * dw;
#pragma unroll
  for (int o = 16; o; o >>= 1) vs += __shfl_xor(vs, o, 32);
  float inv = rsqrtf(vs * (1.0f / 128.0f) + LN_EPS);
  int f = lane * 4;
  float4 gv = *(const float4*)(g + f);
  float4 bv = *(const float4*)(bb + f);
  float4 id = *(const float4*)(hold + base);
  float4 o;
  o.x = fmaxf(dx * inv * gv.x + bv.x, 0.f) + id.x;
  o.y = fmaxf(dy * inv * gv.y + bv.y, 0.f) + id.y;
  o.z = fmaxf(dz * inv * gv.z + bv.z, 0.f) + id.z;
  o.w = fmaxf(dw * inv * gv.w + bv.w, 0.f) + id.w;
  *(float4*)(hnew + base) = o;
}

// ---------------------------------------------------------------------------
extern "C" void kernel_launch(void* const* d_in, const int* in_sizes, int n_in,
                              void* d_out, int out_size, void* d_ws, size_t ws_size,
                              hipStream_t stream) {
  const float*     x      = (const float*)d_in[0];
  const long long* ei     = (const long long*)d_in[1];   // int64 [2,E]
  const float*     w_in   = (const float*)d_in[2];
  const float*     b_in   = (const float*)d_in[3];
  const float*     conv_w = (const float*)d_in[4];       // [L,H,H]
  const float*     conv_b = (const float*)d_in[5];       // [L,H]
  const float*     ln_g   = (const float*)d_in[6];
  const float*     ln_b   = (const float*)d_in[7];
  const float*     w_out  = (const float*)d_in[8];
  const float*     b_out  = (const float*)d_in[9];
  float*           out    = (float*)d_out;

  const int N = in_sizes[0] / 256;   // 50000
  const int E = in_sizes[1] / 2;     // 800000
  const int DIN = 256, DOUT = 32, L = 3;

  auto aup = [](size_t v) { return (v + 255) & ~(size_t)255; };
  char* p = (char*)d_ws;
  float* dis = (float*)p; p += aup((size_t)N * 4);
  float* hA  = (float*)p; p += aup((size_t)N * Hdim * 4);
  float* hB  = (float*)p; p += aup((size_t)N * Hdim * 4);
  float* m   = (float*)p; p += aup((size_t)N * Hdim * 4);
  float* agg = (float*)p; p += aup((size_t)N * Hdim * 4);
  (void)ws_size; (void)n_in; (void)out_size;

  const int nBlkN    = (N + 255) / 256;
  const int nBlkE    = (E + 255) / 256;
  const int nBlkNH   = (N * Hdim + 255) / 256;
  const int nBlkRow  = (N + 127) / 128;
  const int nBlkEdge = (E + 7) / 8;
  const int nBlkNode = (N + 7) / 8;

  // 1) symmetric normalization: deg -> dis = rsqrt(deg+1)
  k_fill0<<<nBlkN, 256, 0, stream>>>(dis, N);
  k_degree<<<nBlkE, 256, 0, stream>>>(ei, dis, E);
  k_rsqrt<<<nBlkN, 256, 0, stream>>>(dis, N);

  // 2) h = relu(x @ w_in + b_in)    K=256 (KC=8), NOUT=128 (NT=8)
  gemm_bf16_wmma<8, 8, true, true>
      <<<nBlkRow, 256, (size_t)DIN * Hdim * 2, stream>>>(x, w_in, b_in, hA, N);

  // 3) three GCN layers
  float* h  = hA;
  float* hN = hB;
  for (int i = 0; i < L; ++i) {
    // m = h @ conv_w[i]            K=128 (KC=4), NOUT=128 (NT=8)
    gemm_bf16_wmma<4, 8, false, false>
        <<<nBlkRow, 256, (size_t)Hdim * Hdim * 2, stream>>>(
            h, conv_w + (size_t)i * Hdim * Hdim, nullptr, m, N);
    // agg = m*self_w + conv_b, then scatter-add neighbors
    k_selfbias<<<nBlkNH, 256, 0, stream>>>(m, dis, conv_b + i * Hdim, agg, N * Hdim);
    k_scatter<<<nBlkEdge, 256, 0, stream>>>(ei, dis, m, agg, E);
    // h = relu(LN(agg)) + h
    k_ln_relu_res<<<nBlkNode, 256, 0, stream>>>(
        agg, ln_g + i * Hdim, ln_b + i * Hdim, h, hN, N);
    float* t = h; h = hN; hN = t;
  }

  // 4) out = h @ w_out + b_out     K=128 (KC=4), NOUT=32 (NT=2)
  gemm_bf16_wmma<4, 2, false, true>
      <<<nBlkRow, 256, (size_t)Hdim * DOUT * 2, stream>>>(h, w_out, b_out, out, N);
}
